// CaSA_11261404250482
// MI455X (gfx1250) — compile-verified, matrix operand order checked by reading, not run on hardware
//
#include <hip/hip_runtime.h>
#include <hip/hip_bf16.h>
#include <math.h>

typedef __attribute__((ext_vector_type(16))) _Float16 v16h;
typedef __attribute__((ext_vector_type(8)))  _Float16 v8h;
typedef __attribute__((ext_vector_type(8)))  float    v8f;
typedef __attribute__((ext_vector_type(4)))  unsigned int u32x4;
typedef __attribute__((ext_vector_type(8)))  int      i32x8;
typedef __attribute__((ext_vector_type(4)))  int      i32x4;

#define DIM   192
#define NPIX  4096
#define BATCH 32
#define HDIM  48
#define ATT_SCALE 0.14433756729740643f  // 48^-0.5

// ---------------------------------------------------------------- helpers ---
__device__ __forceinline__ v8f wmma_f16(v16h a, v16h b, v8f c) {
  // D = A(16x32 f16) * B(32x16 f16) + C(16x16 f32)
  return __builtin_amdgcn_wmma_f32_16x16x32_f16(false, a, false, b, (short)0, c,
                                                false, false);
}

// Load one WMMA A/B fragment from f16 memory laid out with the per-lane
// pattern: elements 0..7 at p[0..7], elements 8..15 at p[16..23].
__device__ __forceinline__ v16h ldfrag(const _Float16* p) {
  v8h lo = *(const v8h*)(p);
  v8h hi = *(const v8h*)(p + 16);
  return __builtin_shufflevector(lo, hi, 0, 1, 2, 3, 4, 5, 6, 7,
                                 8, 9, 10, 11, 12, 13, 14, 15);
}

// Same fragment pattern but source is f32 (weights): 8 floats at p, 8 at p+16.
__device__ __forceinline__ v16h ldfragA_f32(const float* p) {
  float4 a0 = *(const float4*)(p);
  float4 a1 = *(const float4*)(p + 4);
  float4 a2 = *(const float4*)(p + 16);
  float4 a3 = *(const float4*)(p + 20);
  v16h r;
  r[0] = (_Float16)a0.x;  r[1] = (_Float16)a0.y;
  r[2] = (_Float16)a0.z;  r[3] = (_Float16)a0.w;
  r[4] = (_Float16)a1.x;  r[5] = (_Float16)a1.y;
  r[6] = (_Float16)a1.z;  r[7] = (_Float16)a1.w;
  r[8] = (_Float16)a2.x;  r[9] = (_Float16)a2.y;
  r[10] = (_Float16)a2.z; r[11] = (_Float16)a2.w;
  r[12] = (_Float16)a3.x; r[13] = (_Float16)a3.y;
  r[14] = (_Float16)a3.z; r[15] = (_Float16)a3.w;
  return r;
}

// TDM: DMA one (4 x 64 x 48)-element f16 tile (w contiguous, h stride 64,
// channel stride 4096 elements) from global into LDS at lds_off, packed
// [c][h][w] with 4 DWORDs of padding every 128 DWORDs -> row stride 264 f16.
__device__ __forceinline__ void tdm_tile_load(unsigned lds_off,
                                              const void* gptr) {
  const unsigned long long ga = (unsigned long long)(uintptr_t)gptr;
  u32x4 g0;
  g0[0] = 1u;                                       // count = 1 (valid)
  g0[1] = lds_off;                                  // LDS byte address
  g0[2] = (unsigned)(ga & 0xffffffffu);             // global_addr[31:0]
  g0[3] = (unsigned)((ga >> 32) & 0x1ffffffu)       // global_addr[56:32]
          | (2u << 30);                             // type = 2 ("image")
  i32x8 g1;
  g1[0] = (int)((1u << 16)      // data_size = 1 -> 2 bytes
                | (1u << 20)    // pad_enable
                | (6u << 22)    // pad_interval: every 128 DWORDs (512 B)
                | (3u << 25));  // pad_amount: 4 DWORDs (8 f16)
  g1[1] = (int)(4u << 16);      // tensor_dim0 = 4        (bits 79:48)
  g1[2] = (int)(64u << 16);     // tensor_dim1 = 64       (bits 111:80)
  g1[3] = (int)(4u << 16);      // tile_dim0 = 4          (bits 127:112)
  g1[4] = (int)(64u | (48u << 16));  // tile_dim1 = 64, tile_dim2 = 48
  g1[5] = 64;                   // tensor_dim0_stride = 64 elems
  g1[6] = (int)(4096u << 16);   // tensor_dim1_stride = 4096 elems (low 16)
  g1[7] = 0;                    // tensor_dim1_stride high bits = 0
  i32x4 g2 = {48, 0, 0, 0};     // tensor_dim2 = 48 (3D tensor)
  i32x4 g3 = {0, 0, 0, 0};
  i32x8 gz = {0, 0, 0, 0, 0, 0, 0, 0};  // aux group (zero-filled)
  __builtin_amdgcn_tensor_load_to_lds(g0, g1, g2, g3, gz, 0);
}

// ------------------------------------------------------------------- K1 -----
__global__ void __launch_bounds__(256)
k1_qkv_gemm(const float* __restrict__ x,
            const float* __restrict__ w_qk, const float* __restrict__ b_qk,
            const float* __restrict__ w_v,  const float* __restrict__ b_v,
            _Float16* __restrict__ qpre, _Float16* __restrict__ kpre,
            _Float16* __restrict__ vh) {
  __shared__ _Float16 Xs[128 * 200];  // [pixel][k], k-stride padded 192->200
  const int tid = threadIdx.x;
  const int lane = tid & 31, wid = tid >> 5;
  const int p0 = blockIdx.x * 128;
  const int bb = p0 >> 12;        // batch
  const int hw0 = p0 & 4095;      // pixel within image (128 | 4096)

  for (int i = tid; i < DIM * 128; i += 256) {
    int c = i >> 7, p = i & 127;  // coalesced along p
    Xs[p * 200 + c] = (_Float16)x[((size_t)(bb * DIM + c) << 12) + hw0 + p];
  }
  __syncthreads();

  const int wm = wid >> 1, wn = wid & 1;  // 4 M-subtiles x 2 N-halves
  const int n = lane & 15, hi = lane >> 4;
  const int khalf = hi << 3;
  const int pw = wn * 64;

  for (int mt = 0; mt < 9; ++mt) {
    const int ocw = mt * 64 + wm * 16;
    const int oc_row = ocw + n;  // A row for this lane
    const float* wrow = (oc_row < 2 * DIM) ? (w_qk + oc_row * DIM)
                                           : (w_v + (oc_row - 2 * DIM) * DIM);
    v8f acc[4] = {};
    for (int kc = 0; kc < 6; ++kc) {
      const int kb = kc * 32;
      v16h A = ldfragA_f32(wrow + kb + khalf);
#pragma unroll
      for (int nt = 0; nt < 4; ++nt) {
        v16h B = ldfrag(Xs + (pw + nt * 16 + n) * 200 + kb + khalf);
        acc[nt] = wmma_f16(A, B, acc[nt]);
      }
    }
#pragma unroll
    for (int nt = 0; nt < 4; ++nt) {
#pragma unroll
      for (int r = 0; r < 8; ++r) {
        const int oc_o = ocw + r + (hi << 3);  // D row
        _Float16* dst;
        int c;
        float bias;
        if (oc_o < DIM) {
          dst = qpre; c = oc_o; bias = b_qk[oc_o];
        } else if (oc_o < 2 * DIM) {
          dst = kpre; c = oc_o - DIM; bias = b_qk[oc_o];
        } else {
          dst = vh; c = oc_o - 2 * DIM; bias = b_v[oc_o - 2 * DIM];
        }
        dst[((size_t)(bb * DIM + c) << 12) + hw0 + pw + nt * 16 + n] =
            (_Float16)(acc[nt][r] + bias);
      }
    }
  }
}

// ------------------------------------------------------------------- K2 -----
__global__ void __launch_bounds__(256)
k2_dwconv(const _Float16* __restrict__ qpre, const _Float16* __restrict__ kpre,
          const _Float16* __restrict__ vh,
          const float* __restrict__ w_q5, const float* __restrict__ b_q5,
          const float* __restrict__ w_k5, const float* __restrict__ b_k5,
          const float* __restrict__ w_cape, const float* __restrict__ b_cape,
          float* __restrict__ cape, _Float16* __restrict__ qp,
          _Float16* __restrict__ kp) {
  __shared__ _Float16 Tq[400], Tk[400], Tv[400];  // 20x20 halo tiles
  __shared__ float Wq[25], Wk[25], Wc[25];
  const int tid = threadIdx.x;
  int bidx = blockIdx.x;
  const int t = bidx & 15; bidx >>= 4;            // 16 tiles per (b,c)
  const int c = bidx % DIM;
  const int b = bidx / DIM;
  const int h0 = (t >> 2) << 4, w0 = (t & 3) << 4;
  const size_t cbase = (size_t)(b * DIM + c) << 12;

  for (int i = tid; i < 400; i += 256) {
    int hh = i / 20, ww = i % 20;
    int gh = h0 + hh - 2, gw = w0 + ww - 2;
    bool ok = ((unsigned)gh < 64u) && ((unsigned)gw < 64u);
    size_t g = cbase + (gh << 6) + gw;
    Tq[i] = ok ? qpre[g] : (_Float16)0.f;
    Tk[i] = ok ? kpre[g] : (_Float16)0.f;
    Tv[i] = ok ? vh[g] : (_Float16)0.f;
  }
  if (tid < 25) {
    Wq[tid] = w_q5[c * 25 + tid];
    Wk[tid] = w_k5[c * 25 + tid];
    Wc[tid] = w_cape[c * 25 + tid];
  }
  __syncthreads();

  const int ty = tid >> 4, tx = tid & 15;
  float aC = 0.f, aQ = 0.f, aK = 0.f;
#pragma unroll
  for (int i = 0; i < 5; ++i)
#pragma unroll
    for (int j = 0; j < 5; ++j) {
      const int off = (ty + i) * 20 + tx + j;
      aC += Wc[i * 5 + j] * (float)Tv[off];
      aQ += Wq[i * 5 + j] * (float)Tq[off];
      aK += Wk[i * 5 + j] * (float)Tk[off];
    }
  const float cv = aC + b_cape[c];
  const size_t g = cbase + ((h0 + ty) << 6) + (w0 + tx);
  cape[g] = cv;
  qp[g] = (_Float16)(aQ + b_q5[c] + cv);
  kp[g] = (_Float16)(aK + b_k5[c] + cv);
}

// ------------------------------------------------------------------- K3 -----
// One workgroup per (window, head). TDM DMAs Q/K/V tiles to LDS ([d][l],
// row stride 264). V consumed directly (Vt); K transposed LDS->LDS into
// Ks[l][64] (hot B-fragment path, d zero-padded); Q stays [d][l] and its 4
// A-fragments per wave are scalar-gathered. Sm=[256][264] f16 holds S/P;
// Kt scratch aliases Sm; Os (f32 [256][52]) aliases Qt+Ks.
#define QT_OFF 0u
#define KS_OFF (48u * 264u * 2u)                 // 25,344
#define VT_OFF (KS_OFF + 256u * 64u * 2u)        // 58,112
#define SM_OFF (VT_OFF + 48u * 264u * 2u)        // 83,456
#define K3_SMEM (SM_OFF + 256u * 264u * 2u)      // 218,624 bytes

__global__ void __launch_bounds__(256)
k3_attn(const _Float16* __restrict__ qp, const _Float16* __restrict__ kp,
        const _Float16* __restrict__ vh, const float* __restrict__ cape,
        float* __restrict__ out) {
  extern __shared__ char smem[];
  _Float16* Qt = (_Float16*)(smem + QT_OFF);  // [48][264] from TDM
  _Float16* Ks = (_Float16*)(smem + KS_OFF);  // [256][64]
  _Float16* Vt = (_Float16*)(smem + VT_OFF);  // [48][264] from TDM
  _Float16* Sm = (_Float16*)(smem + SM_OFF);  // [256][264]
  _Float16* Kt = Sm;                          // TDM K scratch (aliases Sm)
  float* Os = (float*)smem;                   // [256][52] f32, aliases Qt+Ks

  const int tid = threadIdx.x, lane = tid & 31, wid = tid >> 5;
  const int n = lane & 15, hi = lane >> 4, khalf = hi << 3;
  const int head = blockIdx.x & 3;
  const int win = blockIdx.x >> 2;
  const int b = win >> 4, wx = win & 15;
  const size_t base = (size_t)(b * DIM + head * HDIM) << 12;
  const int pixb = wx << 2;

  // ---- stage Q, K, V via the Tensor Data Mover ----
  if (wid == 0) {
    tdm_tile_load(QT_OFF, qp + base + pixb);
    tdm_tile_load(SM_OFF, kp + base + pixb);   // K -> Kt scratch
    tdm_tile_load(VT_OFF, vh + base + pixb);
    __builtin_amdgcn_s_wait_tensorcnt(0);
  }
  __syncthreads();

  // K: LDS->LDS transpose [d][l] -> [l][d], plus zero-pad d 48..63
  for (int i = tid; i < 256 * HDIM; i += 256) {
    int l = i & 255, d = i >> 8;
    Ks[l * 64 + d] = Kt[d * 264 + l];
  }
  for (int i = tid; i < 256 * 16; i += 256) {
    int l = i & 255, d = 48 + (i >> 8);
    Ks[l * 64 + d] = (_Float16)0.f;
  }
  __syncthreads();

  // ---- S = scale * Q K^T (each wave: 2 row-tiles x 16 col-tiles) ----
  const int mt0 = wid * 2;
  for (int mi = 0; mi < 2; ++mi) {
    const int mrow = (mt0 + mi) * 16 + n;
    v16h A0, A1;
#pragma unroll
    for (int e = 0; e < 8; ++e) {
      A0[e] = Qt[(khalf + e) * 264 + mrow];            // d 0..15
      A0[e + 8] = Qt[(khalf + 16 + e) * 264 + mrow];   // d 16..31
      int d0 = 32 + khalf + e;                         // 32..47 valid
      A1[e] = (d0 < HDIM) ? Qt[d0 * 264 + mrow] : (_Float16)0.f;
      A1[e + 8] = (_Float16)0.f;                       // d >= 48 -> zero pad
    }
    for (int nt = 0; nt < 16; ++nt) {
      const _Float16* krow = Ks + (nt * 16 + n) * 64;
      v16h B0 = ldfrag(krow + khalf);
      v16h B1 = ldfrag(krow + 32 + khalf);
      v8f acc = {};
      acc = wmma_f16(A0, B0, acc);
      acc = wmma_f16(A1, B1, acc);
      _Float16* srow = Sm + ((mt0 + mi) * 16 + (hi << 3)) * 264 + nt * 16 + n;
#pragma unroll
      for (int r = 0; r < 8; ++r) srow[r * 264] = (_Float16)(acc[r] * ATT_SCALE);
    }
  }
  __syncthreads();

  // ---- softmax, one row per thread ----
  {
    _Float16* sr = Sm + tid * 264;
    float mx = -1e30f;
    for (int j = 0; j < 256; ++j) mx = fmaxf(mx, (float)sr[j]);
    float sum = 0.f;
    for (int j = 0; j < 256; ++j) {
      float e = __expf((float)sr[j] - mx);
      sum += e;
      sr[j] = (_Float16)e;
    }
    float inv = 1.f / sum;
    for (int j = 0; j < 256; ++j) sr[j] = (_Float16)((float)sr[j] * inv);
  }
  __syncthreads();

  // ---- O = P V (each wave: 2 row-tiles x 3 col-tiles, K = 256) ----
  for (int mi = 0; mi < 2; ++mi) {
    const _Float16* prow = Sm + ((mt0 + mi) * 16 + n) * 264;
#pragma unroll
    for (int nt = 0; nt < 3; ++nt) {
      const _Float16* vcol = Vt + (nt * 16 + n) * 264;
      v8f acc = {};
#pragma unroll
      for (int kc = 0; kc < 8; ++kc) {
        v16h A = ldfrag(prow + kc * 32 + khalf);
        v16h B = ldfrag(vcol + kc * 32 + khalf);
        acc = wmma_f16(A, B, acc);
      }
      float* orow = Os + ((mt0 + mi) * 16 + (hi << 3)) * 52 + nt * 16 + n;
#pragma unroll
      for (int r = 0; r < 8; ++r) orow[r * 52] = acc[r];
    }
  }
  __syncthreads();

  // ---- write out + cape, coalesced float4 along w ----
  for (int i = tid; i < HDIM * 64; i += 256) {
    int d = i >> 6, h = i & 63;
    size_t g = base + ((size_t)d << 12) + (h << 6) + pixb;
    float4 cv = *(const float4*)(cape + g);
    int l0 = h << 2;
    float4 ov;
    ov.x = Os[(l0 + 0) * 52 + d] + cv.x;
    ov.y = Os[(l0 + 1) * 52 + d] + cv.y;
    ov.z = Os[(l0 + 2) * 52 + d] + cv.z;
    ov.w = Os[(l0 + 3) * 52 + d] + cv.w;
    *(float4*)(out + g) = ov;
  }
}

// ------------------------------------------------------------- launcher -----
extern "C" void kernel_launch(void* const* d_in, const int* in_sizes, int n_in,
                              void* d_out, int out_size, void* d_ws,
                              size_t ws_size, hipStream_t stream) {
  (void)in_sizes; (void)n_in; (void)out_size; (void)ws_size;
  const float* x      = (const float*)d_in[0];
  const float* w_qk   = (const float*)d_in[1];
  const float* b_qk   = (const float*)d_in[2];
  const float* w_q5   = (const float*)d_in[3];
  const float* b_q5   = (const float*)d_in[4];
  const float* w_k5   = (const float*)d_in[5];
  const float* b_k5   = (const float*)d_in[6];
  const float* w_v    = (const float*)d_in[7];
  const float* b_v    = (const float*)d_in[8];
  const float* w_cape = (const float*)d_in[9];
  const float* b_cape = (const float*)d_in[10];
  float* out = (float*)d_out;

  char* ws = (char*)d_ws;
  const size_t NELT = (size_t)BATCH * DIM * NPIX;  // 25,165,824 elements
  _Float16* qpre = (_Float16*)(ws);                // f16, 2*NELT bytes
  _Float16* kpre = (_Float16*)(ws + 2 * NELT);
  _Float16* vh   = (_Float16*)(ws + 4 * NELT);
  _Float16* qp   = (_Float16*)(ws + 6 * NELT);
  _Float16* kp   = (_Float16*)(ws + 8 * NELT);
  float*    cape = (float*)(ws + 10 * NELT);       // f32, 4*NELT bytes

  k1_qkv_gemm<<<1024, 256, 0, stream>>>(x, w_qk, b_qk, w_v, b_v, qpre, kpre, vh);
  k2_dwconv<<<BATCH * DIM * 16, 256, 0, stream>>>(qpre, kpre, vh, w_q5, b_q5,
                                                  w_k5, b_k5, w_cape, b_cape,
                                                  cape, qp, kp);
  (void)hipFuncSetAttribute((const void*)k3_attn,
                            hipFuncAttributeMaxDynamicSharedMemorySize,
                            (int)K3_SMEM);
  k3_attn<<<2048, 256, K3_SMEM, stream>>>(qp, kp, vh, cape, out);
}
